// GraphTransformerModel_32263794327581
// MI455X (gfx1250) — matrix operand order repaired and checked by reading, not compiled, not run on hardware
//
#include <hip/hip_runtime.h>
#include <cmath>

typedef __attribute__((ext_vector_type(16))) _Float16 v16h;
typedef __attribute__((ext_vector_type(8)))  _Float16 v8h;
typedef __attribute__((ext_vector_type(4)))  _Float16 v4h;
typedef __attribute__((ext_vector_type(8)))  float    v8f;

union AFrag { v16h v; v8h h[2]; };

// ---------------------------------------------------------------- utilities
__global__ void fill_i32(int* __restrict__ p, int v, long n) {
    long i = (long)blockIdx.x * blockDim.x + threadIdx.x;
    if (i < n) p[i] = v;
}

__global__ void copy_f32(const float* __restrict__ s, float* __restrict__ d, long n) {
    long i = (long)blockIdx.x * blockDim.x + threadIdx.x;
    if (i < n) d[i] = s[i];
}

__global__ void cvt_f32_f16(const float* __restrict__ s, _Float16* __restrict__ d, long n) {
    long i = (long)blockIdx.x * blockDim.x + threadIdx.x;
    if (i < n) d[i] = (_Float16)s[i];
}

// W [K, Ncols] f32 row-major  ->  WT [Ncols, K] f16 row-major
__global__ void wtrans_f16(const float* __restrict__ W, _Float16* __restrict__ WT,
                           int K, int Ncols) {
    long i = (long)blockIdx.x * blockDim.x + threadIdx.x;
    if (i >= (long)K * Ncols) return;
    int k = (int)(i / Ncols), n = (int)(i % Ncols);
    WT[(size_t)n * K + k] = (_Float16)W[i];
}

// ---------------------------------------------------------------- WMMA GEMM
// C[M, Ncols] = op(A[M,K] @ Bt[Ncols,K]^T + bias), A/Bt f16, f32 accumulate.
// One wave computes a 16x64 strip. blockDim = 128 (4 waves -> 64 rows/block).
__global__ void gemm_wmma(const _Float16* __restrict__ A,
                          const _Float16* __restrict__ Bt,
                          const float* __restrict__ bias,
                          _Float16* __restrict__ C16,
                          int M, int K, int Ncols, int doRelu) {
    int wid  = threadIdx.x >> 5;
    int lane = threadIdx.x & 31;
    int hi   = lane >> 4;      // 0: lanes 0-15, 1: lanes 16-31
    int l15  = lane & 15;
    int m0   = (blockIdx.x * 4 + wid) * 16;
    int n0   = blockIdx.y * 64;

    v8f acc0 = {}, acc1 = {}, acc2 = {}, acc3 = {};

    const _Float16* arow = A + (size_t)(m0 + l15) * K;
    const _Float16* brow = Bt + (size_t)(n0 + l15) * K;

    for (int kb = 0; kb < K; kb += 32) {
        // A fragment: lane holds row m0+l15; K = kb + hi*16 + 0..15 contiguous
        AFrag af;
        int ka = kb + hi * 16;
        af.h[0] = *reinterpret_cast<const v8h*>(arow + ka);
        af.h[1] = *reinterpret_cast<const v8h*>(arow + ka + 8);
        // B fragments: lane holds col (n-tile)+l15; K = kb + hi*16 + 0..15 contiguous
        const _Float16* bb = brow + kb + hi * 16;
        v16h b0 = *reinterpret_cast<const v16h*>(bb);
        v16h b1 = *reinterpret_cast<const v16h*>(bb + (size_t)16 * K);
        v16h b2 = *reinterpret_cast<const v16h*>(bb + (size_t)32 * K);
        v16h b3 = *reinterpret_cast<const v16h*>(bb + (size_t)48 * K);
        acc0 = __builtin_amdgcn_wmma_f32_16x16x32_f16(false, af.v, false, b0, (short)0, acc0, false, false);
        acc1 = __builtin_amdgcn_wmma_f32_16x16x32_f16(false, af.v, false, b1, (short)0, acc1, false, false);
        acc2 = __builtin_amdgcn_wmma_f32_16x16x32_f16(false, af.v, false, b2, (short)0, acc2, false, false);
        acc3 = __builtin_amdgcn_wmma_f32_16x16x32_f16(false, af.v, false, b3, (short)0, acc3, false, false);
    }

    v8f accs[4] = {acc0, acc1, acc2, acc3};
#pragma unroll
    for (int t = 0; t < 4; ++t) {
        int col = n0 + t * 16 + l15;
        float bv = bias ? bias[col] : 0.0f;
#pragma unroll
        for (int r = 0; r < 8; ++r) {
            int row = m0 + r + hi * 8;   // C layout: VGPR r -> M=r (lanes<16) / M=8+r
            if (row < M) {
                float val = accs[t][r] + bv;
                if (doRelu) val = fmaxf(val, 0.0f);
                C16[(size_t)row * Ncols + col] = (_Float16)val;
            }
        }
    }
}

// ---------------------------------------------------------------- CSR build
__global__ void deg_count(const int* __restrict__ dst, int* __restrict__ deg, int E) {
    int e = blockIdx.x * blockDim.x + threadIdx.x;
    if (e < E) atomicAdd(&deg[dst[e]], 1);
}

// Block-level exclusive scan (1024 elements per block) + block sums.
__global__ void scan_block(const int* __restrict__ deg, int* __restrict__ part,
                           int* __restrict__ bsums, int n) {
    __shared__ int sh[1024];
    int tid = threadIdx.x;
    long i = (long)blockIdx.x * 1024 + tid;
    int v = (i < n) ? deg[i] : 0;
    sh[tid] = v;
    __syncthreads();
    for (int off = 1; off < 1024; off <<= 1) {
        int t = (tid >= off) ? sh[tid - off] : 0;
        __syncthreads();
        sh[tid] += t;
        __syncthreads();
    }
    int incl = sh[tid];
    if (i < n) part[i] = incl - v;
    if (tid == 1023) bsums[blockIdx.x] = incl;
}

__global__ void scan_sums(int* __restrict__ bsums, int nb) {
    if (threadIdx.x == 0 && blockIdx.x == 0) {
        int run = 0;
        for (int b = 0; b < nb; ++b) { int t = bsums[b]; bsums[b] = run; run += t; }
    }
}

__global__ void scan_add(const int* __restrict__ part, const int* __restrict__ bsums,
                         int* __restrict__ rowptr, int* __restrict__ cursor, int n, int E) {
    long i = (long)blockIdx.x * blockDim.x + threadIdx.x;
    if (i < n) {
        int v = part[i] + bsums[i / 1024];
        rowptr[i] = v;
        cursor[i] = v;
    }
    if (i == 0) rowptr[n] = E;
}

__global__ void csr_scatter(const int* __restrict__ dst, int* __restrict__ cursor,
                            int* __restrict__ elist, int E) {
    int e = blockIdx.x * blockDim.x + threadIdx.x;
    if (e >= E) return;
    int p = atomicAdd(&cursor[dst[e]], 1);
    elist[p] = e;
}

// ---------------------------------------------------------------- attention
// Scores only: scores[e,h] = (q[dst[e]] . k[src[e]]) * scale. One wave per edge.
__global__ void edge_scores(const _Float16* __restrict__ q, const _Float16* __restrict__ k,
                            int stride, const int* __restrict__ src, const int* __restrict__ dst,
                            float* __restrict__ scores, int E, int heads, float scale) {
    int wid = threadIdx.x >> 5, lane = threadIdx.x & 31;
    int e = blockIdx.x * 4 + wid;
    if (e >= E) return;
    const _Float16* qrow = q + (size_t)dst[e] * stride;
    const _Float16* krow = k + (size_t)src[e] * stride;
    for (int h = 0; h < heads; ++h) {
        int c0 = h * 128 + lane * 4;
        v4h qv = *reinterpret_cast<const v4h*>(qrow + c0);
        v4h kv = *reinterpret_cast<const v4h*>(krow + c0);
        float acc = 0.0f;
#pragma unroll
        for (int j = 0; j < 4; ++j) acc += (float)qv[j] * (float)kv[j];
#pragma unroll
        for (int off = 16; off >= 1; off >>= 1) acc += __shfl_xor(acc, off, 32);
        if (lane == 0) scores[(size_t)e * heads + h] = acc * scale;
    }
}

// Gather-based aggregation: one wave per destination node. Walks the node's
// in-edge list twice (max, then exp/weighted-sum), all in registers, fuses
// softmax-normalize + skip + ReLU. Atomic-free and deterministic.
template <int NH>
__global__ void node_aggregate(const _Float16* __restrict__ qkvs, int stride, int voff, int soff,
                               const int* __restrict__ rowptr, const int* __restrict__ elist,
                               const int* __restrict__ src, const float* __restrict__ scores,
                               _Float16* __restrict__ o16, float* __restrict__ o32,
                               int outStride, int Nn) {
    int wid = threadIdx.x >> 5, lane = threadIdx.x & 31;
    int nid = blockIdx.x * 4 + wid;
    if (nid >= Nn) return;
    int t0 = rowptr[nid], t1 = rowptr[nid + 1];

    float mx[NH];
#pragma unroll
    for (int h = 0; h < NH; ++h) mx[h] = -__builtin_inff();
    for (int t = t0; t < t1; ++t) {
        int e = elist[t];
#pragma unroll
        for (int h = 0; h < NH; ++h) mx[h] = fmaxf(mx[h], scores[(size_t)e * NH + h]);
    }

    float den[NH];
    float acc[NH * 4];
#pragma unroll
    for (int h = 0; h < NH; ++h) den[h] = 0.0f;
#pragma unroll
    for (int i = 0; i < NH * 4; ++i) acc[i] = 0.0f;

    for (int t = t0; t < t1; ++t) {
        int e = elist[t];
        const _Float16* vrow = qkvs + (size_t)src[e] * stride + voff;
#pragma unroll
        for (int h = 0; h < NH; ++h) {
            float eh = __expf(scores[(size_t)e * NH + h] - mx[h]);
            den[h] += eh;
            v4h vv = *reinterpret_cast<const v4h*>(vrow + h * 128 + lane * 4);
#pragma unroll
            for (int j = 0; j < 4; ++j) acc[h * 4 + j] += eh * (float)vv[j];
        }
    }

    const _Float16* srow = qkvs + (size_t)nid * stride + soff;
#pragma unroll
    for (int h = 0; h < NH; ++h) {
        float inv = 1.0f / (den[h] + 1e-16f);
#pragma unroll
        for (int j = 0; j < 4; ++j) {
            int c = h * 128 + lane * 4 + j;
            float val = acc[h * 4 + j] * inv + (float)srow[c];
            val = fmaxf(val, 0.0f);
            size_t idx = (size_t)nid * outStride + c;
            if (o16) o16[idx] = (_Float16)val;
            if (o32) o32[idx] = val;
        }
    }
}

// ---------------------------------------------------------------- pooling + heads
// batch is sorted: per-graph range via binary search; deterministic reduction.
__global__ void pool_mean(const float* __restrict__ h2, const int* __restrict__ batch,
                          float* __restrict__ g, int Nn) {
    int b = blockIdx.x, j = threadIdx.x;   // 64 blocks x 128 threads
    int lo = 0, hi = Nn;
    while (lo < hi) { int mid = (lo + hi) >> 1; if (batch[mid] < b) lo = mid + 1; else hi = mid; }
    int s = lo;
    lo = 0; hi = Nn;
    while (lo < hi) { int mid = (lo + hi) >> 1; if (batch[mid] < b + 1) lo = mid + 1; else hi = mid; }
    int e = lo;
    float acc = 0.0f;
    for (int i = s; i < e; ++i) acc += h2[(size_t)i * 128 + j];
    g[b * 128 + j] = acc / fmaxf((float)(e - s), 1.0f);
}

__global__ void head_linear(const float* __restrict__ g, const float* __restrict__ W,
                            const float* __restrict__ bias, float* __restrict__ out, int Wd) {
    int idx = blockIdx.x * blockDim.x + threadIdx.x;
    if (idx >= 64 * Wd) return;
    int b = idx / Wd, w = idx % Wd;
    float acc = bias[w];
    const float* gr = g + b * 128;
#pragma unroll 8
    for (int j = 0; j < 128; ++j) acc += gr[j] * W[j * Wd + w];
    out[idx] = acc;
}

// ---------------------------------------------------------------- launcher
extern "C" void kernel_launch(void* const* d_in, const int* in_sizes, int n_in,
                              void* d_out, int out_size, void* d_ws, size_t ws_size,
                              hipStream_t stream) {
    (void)n_in; (void)out_size; (void)ws_size;
    const int N = in_sizes[2];          // batch array length = #nodes
    const int E = in_sizes[1] / 2;
    const int F = 64, H = 128, HEADS = 4, B = 64;
    const int HC1 = HEADS * H;          // 512
    const int Mpad = ((N + 63) / 64) * 64;

    const float* x_f32 = (const float*)d_in[0];
    const int*   src   = (const int*)d_in[1];
    const int*   dst   = src + E;
    const int*   batch = (const int*)d_in[2];

    // workspace layout
    char* base = (char*)d_ws;
    size_t off = 0;
    auto alloc = [&](size_t bytes) -> void* {
        void* p = base + off;
        off = (off + bytes + 255) & ~(size_t)255;
        return p;
    };
    _Float16* xpad    = (_Float16*)alloc((size_t)Mpad * F * 2);
    _Float16* WpT     = (_Float16*)alloc((size_t)H * F * 2);
    _Float16* h1      = (_Float16*)alloc((size_t)Mpad * H * 2);
    _Float16* W1Tcat  = (_Float16*)alloc((size_t)4 * HC1 * H * 2);       // [2048][128]
    float*    bcat1   = (float*)alloc((size_t)4 * HC1 * 4);              // [2048]
    _Float16* qkvs1   = (_Float16*)alloc((size_t)Mpad * 4 * HC1 * 2);    // row: q|k|v|s (each 512)
    float*    scores1 = (float*)alloc((size_t)E * HEADS * 4);
    _Float16* out1    = (_Float16*)alloc((size_t)Mpad * HC1 * 2);
    _Float16* W2Tcat  = (_Float16*)alloc((size_t)4 * H * HC1 * 2);       // [512][512]
    float*    bcat2   = (float*)alloc((size_t)4 * H * 4);                // [512]
    _Float16* qkvs2   = (_Float16*)alloc((size_t)Mpad * 4 * H * 2);      // row: q|k|v|s (each 128)
    float*    scores2 = (float*)alloc((size_t)E * 4);
    float*    h2      = (float*)alloc((size_t)Mpad * H * 4);
    float*    g       = (float*)alloc((size_t)B * H * 4);
    int*      deg     = (int*)alloc((size_t)N * 4);
    int*      part    = (int*)alloc((size_t)N * 4);
    int*      bsums   = (int*)alloc(256 * 4);
    int*      rowptr  = (int*)alloc((size_t)(N + 1) * 4);
    int*      cursor  = (int*)alloc((size_t)N * 4);
    int*      elist   = (int*)alloc((size_t)E * 4);

    auto blocks = [](long n, int bs) { return dim3((unsigned)((n + bs - 1) / bs)); };

    // ---- weight conversion (f32 -> transposed f16, Q|K|V|S concatenated)
    {
        long n = (long)N * F;
        cvt_f32_f16<<<blocks(n, 256), dim3(256), 0, stream>>>(x_f32, xpad, n);
        long nw = (long)F * H;
        wtrans_f16<<<blocks(nw, 256), dim3(256), 0, stream>>>((const float*)d_in[3], WpT, F, H);
        for (int i = 0; i < 4; ++i) {
            long n1 = (long)H * HC1;
            wtrans_f16<<<blocks(n1, 256), dim3(256), 0, stream>>>(
                (const float*)d_in[5 + 2 * i], W1Tcat + (size_t)i * HC1 * H, H, HC1);
            copy_f32<<<blocks(HC1, 256), dim3(256), 0, stream>>>(
                (const float*)d_in[6 + 2 * i], bcat1 + (size_t)i * HC1, HC1);
            long n2 = (long)HC1 * H;
            wtrans_f16<<<blocks(n2, 256), dim3(256), 0, stream>>>(
                (const float*)d_in[13 + 2 * i], W2Tcat + (size_t)i * H * HC1, HC1, H);
            copy_f32<<<blocks(H, 256), dim3(256), 0, stream>>>(
                (const float*)d_in[14 + 2 * i], bcat2 + (size_t)i * H, H);
        }
    }

    // ---- CSR of in-edges (dst -> list of edge ids)
    {
        const int nb = (N + 1023) / 1024;
        fill_i32<<<blocks(N, 256), dim3(256), 0, stream>>>(deg, 0, N);
        deg_count<<<blocks(E, 256), dim3(256), 0, stream>>>(dst, deg, E);
        scan_block<<<dim3((unsigned)nb), dim3(1024), 0, stream>>>(deg, part, bsums, N);
        scan_sums<<<dim3(1), dim3(32), 0, stream>>>(bsums, nb);
        scan_add<<<blocks(N, 256), dim3(256), 0, stream>>>(part, bsums, rowptr, cursor, N, E);
        csr_scatter<<<blocks(E, 256), dim3(256), 0, stream>>>(dst, cursor, elist, E);
    }

    const float scale = 1.0f / sqrtf((float)H);
    const dim3  gemmBlk(128);

    // ---- initial projection: h1 = relu(x @ Wp + bp)
    gemm_wmma<<<dim3(Mpad / 64, H / 64), gemmBlk, 0, stream>>>(
        xpad, WpT, (const float*)d_in[4], h1, N, F, H, 1);

    // ---- conv1: fused Q|K|V|S projection, Ncols = 2048
    gemm_wmma<<<dim3(Mpad / 64, (4 * HC1) / 64), gemmBlk, 0, stream>>>(
        h1, W1Tcat, bcat1, qkvs1, N, H, 4 * HC1, 0);

    // ---- conv1 attention (4 heads)
    edge_scores<<<blocks(E, 4), dim3(128), 0, stream>>>(
        qkvs1 /*q*/, qkvs1 + HC1 /*k*/, 4 * HC1, src, dst, scores1, E, HEADS, scale);
    node_aggregate<4><<<blocks(N, 4), dim3(128), 0, stream>>>(
        qkvs1, 4 * HC1, 2 * HC1 /*voff*/, 3 * HC1 /*soff*/, rowptr, elist, src, scores1,
        out1, nullptr, HC1, N);

    // ---- conv2: fused Q|K|V|S projection, K = 512, Ncols = 512
    gemm_wmma<<<dim3(Mpad / 64, (4 * H) / 64), gemmBlk, 0, stream>>>(
        out1, W2Tcat, bcat2, qkvs2, N, HC1, 4 * H, 0);

    // ---- conv2 attention (1 head)
    edge_scores<<<blocks(E, 4), dim3(128), 0, stream>>>(
        qkvs2, qkvs2 + H, 4 * H, src, dst, scores2, E, 1, scale);
    node_aggregate<1><<<blocks(N, 4), dim3(128), 0, stream>>>(
        qkvs2, 4 * H, 2 * H, 3 * H, rowptr, elist, src, scores2,
        nullptr, h2, H, N);

    // ---- global mean pool (deterministic, atomic-free)
    pool_mean<<<dim3((unsigned)B), dim3(128), 0, stream>>>(h2, batch, g, N);

    // ---- output heads (concatenated flat in return order)
    struct HeadSpec { int wi, bi, off, wd; };
    const HeadSpec hs[7] = {
        {21, 22, 0,     1},
        {23, 24, 64,    4},
        {25, 26, 320,   3},
        {27, 28, 512,   512},
        {29, 30, 33280, 512},
        {31, 32, 66048, 512},
        {33, 34, 98816, 8},
    };
    float* out = (float*)d_out;
    for (int i = 0; i < 7; ++i) {
        int n = 64 * hs[i].wd;
        head_linear<<<blocks(n, 256), dim3(256), 0, stream>>>(
            g, (const float*)d_in[hs[i].wi], (const float*)d_in[hs[i].bi], out + hs[i].off, hs[i].wd);
    }
}
